// MoE_16252156248245
// MI455X (gfx1250) — matrix operand order repaired and checked by reading, not compiled, not run on hardware
//
#include <hip/hip_runtime.h>
#include <math.h>

#define DIM    1024
#define INTER  512
#define NEXP   16
#define TOPK   4
#define TOKENS 1024   // 2 * 512

typedef __attribute__((ext_vector_type(16))) __bf16 v16bf;
typedef __attribute__((ext_vector_type(2)))  __bf16 v2bf;
typedef __attribute__((ext_vector_type(8)))  float  v8f;

union BFrag { v16bf v; unsigned int u[8]; };

// packed f32x2 -> bf16x2 : hardware v_cvt_pk_bf16_f32 (1 VALU op per pair)
__device__ __forceinline__ unsigned int f2bf_pk(float lo, float hi) {
#if __has_builtin(__builtin_amdgcn_cvt_pk_bf16_f32)
  union { v2bf v; unsigned int u; } cv;
  cv.v = __builtin_amdgcn_cvt_pk_bf16_f32(lo, hi);
  return cv.u;
#else
  unsigned int r;
  asm("v_cvt_pk_bf16_f32 %0, %1, %2" : "=v"(r) : "v"(lo), "v"(hi));
  return r;
#endif
}

__device__ __forceinline__ v8f wmma_bf16(const BFrag& a, const BFrag& b, v8f c) {
  return __builtin_amdgcn_wmma_f32_16x16x32_bf16(false, a.v, false, b.v,
                                                 (short)0, c, false, false);
}

// ---------------------------------------------------------------------------
// Shared pipelined K-loop: A row (gathered token / H row) x 8 column tiles of
// B (stride LDB between columns, col tile j at +j*128*LDB). Raw B float4s are
// double-buffered across j so ~8 loads stay in flight under each convert+WMMA,
// and next-K lines are prefetched (global_prefetch_b8).
// ---------------------------------------------------------------------------
template<int K, int LDB>
__device__ __forceinline__ void gemm_core(const float* __restrict__ arow,
                                          const float* __restrict__ bcol0,
                                          int hi, v8f acc[8])
{
  for (int kk = 0; kk < K; kk += 32) {
    // ---- A fragment (16x32 bf16, contiguous k = 16*hi + s) ----
    BFrag a;
    {
      const float4* pa = (const float4*)(arow + kk + hi * 16);
      float4 a0 = pa[0], a1 = pa[1];
      const float4* qa = (const float4*)(arow + kk + hi * 16 + 8);
      float4 a2 = qa[0], a3 = qa[1];
      a.u[0] = f2bf_pk(a0.x, a0.y); a.u[1] = f2bf_pk(a0.z, a0.w);
      a.u[2] = f2bf_pk(a1.x, a1.y); a.u[3] = f2bf_pk(a1.z, a1.w);
      a.u[4] = f2bf_pk(a2.x, a2.y); a.u[5] = f2bf_pk(a2.z, a2.w);
      a.u[6] = f2bf_pk(a3.x, a3.y); a.u[7] = f2bf_pk(a3.z, a3.w);
    }
    __builtin_prefetch(arow + kk + 32 + hi * 8, 0, 1);   // next-K A line

    const float* bp = bcol0 + kk + hi * 16;

    // prologue: raw B for j = 0
    float4 c0, c1, c2, c3;
    {
      const float4* p = (const float4*)bp;
      c0 = p[0]; c1 = p[1]; c2 = p[2]; c3 = p[3];
    }
    #pragma unroll
    for (int j = 0; j < 8; ++j) {
      float4 n0, n1, n2, n3;
      if (j < 7) {   // issue j+1 loads before consuming j
        const float4* q = (const float4*)(bp + (size_t)(j + 1) * 128 * LDB);
        n0 = q[0]; n1 = q[1]; n2 = q[2]; n3 = q[3];
      }
      __builtin_prefetch(bp + (size_t)j * 128 * LDB + 32, 0, 1);  // next-K B line
      BFrag b;
      b.u[0] = f2bf_pk(c0.x, c0.y); b.u[1] = f2bf_pk(c0.z, c0.w);
      b.u[2] = f2bf_pk(c1.x, c1.y); b.u[3] = f2bf_pk(c1.z, c1.w);
      b.u[4] = f2bf_pk(c2.x, c2.y); b.u[5] = f2bf_pk(c2.z, c2.w);
      b.u[6] = f2bf_pk(c3.x, c3.y); b.u[7] = f2bf_pk(c3.z, c3.w);
      acc[j] = wmma_bf16(a, b, acc[j]);
      c0 = n0; c1 = n1; c2 = n2; c3 = n3;   // SSA rotation under full unroll
    }
  }
}

// ---------------------------------------------------------------------------
// Kernel 0: zero output
// ---------------------------------------------------------------------------
__global__ void zero_f32(float* __restrict__ p, int n) {
  int i = blockIdx.x * blockDim.x + threadIdx.x;
  if (i < n) p[i] = 0.f;
}

// ---------------------------------------------------------------------------
// Kernel 1: gating. One wave per token. comb[t][e] = routed combine weight.
// ---------------------------------------------------------------------------
__global__ void __launch_bounds__(256)
moe_gate(const float* __restrict__ x,       // [T, DIM]
         const float* __restrict__ gw,      // [NEXP, DIM]
         const float* __restrict__ gb,      // [NEXP]
         float* __restrict__ comb)          // [T, NEXP]
{
  const int wave = threadIdx.x >> 5;
  const int lane = threadIdx.x & 31;
  const int t = blockIdx.x * 8 + wave;

  float acc[NEXP];
  #pragma unroll
  for (int e = 0; e < NEXP; ++e) acc[e] = 0.f;

  const float* xr = x + (size_t)t * DIM;
  for (int i = lane; i < DIM; i += 32) {
    float xv = xr[i];
    #pragma unroll
    for (int e = 0; e < NEXP; ++e) acc[e] += xv * gw[e * DIM + i];
  }
  #pragma unroll
  for (int e = 0; e < NEXP; ++e)
    for (int off = 16; off >= 1; off >>= 1)
      acc[e] += __shfl_xor(acc[e], off, 32);

  if (lane == 0) {
    float s[NEXP], b[NEXP];
    #pragma unroll
    for (int e = 0; e < NEXP; ++e) {
      s[e] = 1.f / (1.f + __expf(-acc[e]));
      b[e] = s[e] + gb[e];
    }
    int idx[TOPK];
    bool used[NEXP];
    #pragma unroll
    for (int e = 0; e < NEXP; ++e) used[e] = false;
    for (int k = 0; k < TOPK; ++k) {
      float best = -1e30f; int bi = 0;
      for (int e = 0; e < NEXP; ++e)
        if (!used[e] && b[e] > best) { best = b[e]; bi = e; }
      used[bi] = true; idx[k] = bi;
    }
    float wsum = 0.f;
    for (int k = 0; k < TOPK; ++k) wsum += s[idx[k]];
    float out[NEXP];
    #pragma unroll
    for (int e = 0; e < NEXP; ++e) out[e] = 0.f;
    for (int k = 0; k < TOPK; ++k) out[idx[k]] = s[idx[k]] / wsum * 2.5f;
    #pragma unroll
    for (int e = 0; e < NEXP; ++e) comb[t * NEXP + e] = out[e];
  }
}

// ---------------------------------------------------------------------------
// Kernel 2: deterministic per-expert token lists + exclusive prefix offsets.
// ---------------------------------------------------------------------------
__global__ void moe_lists(const float* __restrict__ comb,
                          int* __restrict__ list,   // [NEXP, TOKENS]
                          int* __restrict__ offs)   // [NEXP+1]
{
  __shared__ int cnts[NEXP];
  const int wave = threadIdx.x >> 5;
  const int lane = threadIdx.x & 31;
  if (lane == 0) {
    int c = 0;
    for (int t = 0; t < TOKENS; ++t)
      if (comb[t * NEXP + wave] > 0.f) list[wave * TOKENS + c++] = t;
    cnts[wave] = c;
  }
  __syncthreads();
  if (threadIdx.x == 0) {
    int o = 0;
    for (int e = 0; e < NEXP; ++e) { offs[e] = o; o += cnts[e]; }
    offs[NEXP] = o;
  }
}

// ---------------------------------------------------------------------------
// Kernel 3: grouped GEMM1 + fused SiLU*up.
// grid = (64 tiles, 17): y<16 routed expert (gathered rows), y==16 shared.
// ---------------------------------------------------------------------------
__global__ void __launch_bounds__(256)
moe_gemm1(const float* __restrict__ x,      // [T, DIM]
          const float* __restrict__ w13,    // [NEXP, 2I, DIM]
          const float* __restrict__ w13s,   // [2I, DIM]
          const int*   __restrict__ list,
          const int*   __restrict__ offs,
          float* __restrict__ Hrt,          // [4096, INTER]
          float* __restrict__ Hsh)          // [TOKENS, INTER]
{
  const int e    = blockIdx.y;
  const int tile = blockIdx.x;
  const int lane = threadIdx.x & 31;
  const int wave = threadIdx.x >> 5;
  const int hi   = lane >> 4;
  const int lm   = lane & 15;

  __shared__ int tlds[16];

  int mlimit = 16;
  const float* wptr;
  float* hout;
  if (e < NEXP) {
    const int o0  = offs[e];
    const int cnt = offs[e + 1] - o0;
    if (tile * 16 >= cnt) return;            // uniform per block
    const int rem = cnt - tile * 16;
    if (rem < 16) mlimit = rem;
    if (threadIdx.x < 16) {
      int s = tile * 16 + (int)threadIdx.x;
      if (s >= cnt) s = cnt - 1;
      tlds[threadIdx.x] = list[e * TOKENS + s];
    }
    wptr = w13 + (size_t)e * (2 * INTER) * DIM;
    hout = Hrt + (size_t)(o0 + tile * 16) * INTER;
  } else {
    if (threadIdx.x < 16) tlds[threadIdx.x] = tile * 16 + (int)threadIdx.x;
    wptr = w13s;
    hout = Hsh + (size_t)(tile * 16) * INTER;
  }
  __syncthreads();

  const float* xrow = x + (size_t)tlds[lm] * DIM;
  const int ncol = wave * 16 + lm;

  const v8f vz = {0.f,0.f,0.f,0.f,0.f,0.f,0.f,0.f};
  v8f acc[8];
  #pragma unroll
  for (int j = 0; j < 8; ++j) acc[j] = vz;

  gemm_core<DIM, DIM>(xrow, wptr + (size_t)ncol * DIM, hi, acc);

  // col tile j (gate, cols 0..511) pairs with j+4 (up, +512 cols)
  #pragma unroll
  for (int j = 0; j < 4; ++j) {
    const int col = ncol + j * 128;
    #pragma unroll
    for (int r = 0; r < 8; ++r) {
      const int m = r + hi * 8;
      if (m < mlimit) {
        float g = acc[j][r];
        float h = (g / (1.f + __expf(-g))) * acc[j + 4][r];
        hout[(size_t)m * INTER + col] = h;
      }
    }
  }
}

// ---------------------------------------------------------------------------
// Kernel 4: grouped GEMM2, scale by combine weight, atomic accumulate into y.
// ---------------------------------------------------------------------------
__global__ void __launch_bounds__(256)
moe_gemm2(const float* __restrict__ Hrt,
          const float* __restrict__ Hsh,
          const float* __restrict__ w2,     // [NEXP, DIM, INTER]
          const float* __restrict__ w2s,    // [DIM, INTER]
          const int*   __restrict__ list,
          const int*   __restrict__ offs,
          const float* __restrict__ comb,   // [T, NEXP]
          float* __restrict__ y)            // [T, DIM]
{
  const int e    = blockIdx.y;
  const int tile = blockIdx.x;
  const int lane = threadIdx.x & 31;
  const int wave = threadIdx.x >> 5;
  const int hi   = lane >> 4;
  const int lm   = lane & 15;

  __shared__ int tlds[16];

  int mlimit = 16;
  const float* hsrc;
  const float* wptr;
  if (e < NEXP) {
    const int o0  = offs[e];
    const int cnt = offs[e + 1] - o0;
    if (tile * 16 >= cnt) return;
    const int rem = cnt - tile * 16;
    if (rem < 16) mlimit = rem;
    if (threadIdx.x < 16) {
      int s = tile * 16 + (int)threadIdx.x;
      if (s >= cnt) s = cnt - 1;
      tlds[threadIdx.x] = list[e * TOKENS + s];
    }
    hsrc = Hrt + (size_t)(o0 + tile * 16) * INTER;
    wptr = w2 + (size_t)e * DIM * INTER;
  } else {
    if (threadIdx.x < 16) tlds[threadIdx.x] = tile * 16 + (int)threadIdx.x;
    hsrc = Hsh + (size_t)(tile * 16) * INTER;
    wptr = w2s;
  }
  __syncthreads();

  const float* hrow = hsrc + (size_t)lm * INTER;
  const int ncol = wave * 16 + lm;

  const v8f vz = {0.f,0.f,0.f,0.f,0.f,0.f,0.f,0.f};
  v8f acc[8];
  #pragma unroll
  for (int j = 0; j < 8; ++j) acc[j] = vz;

  gemm_core<INTER, INTER>(hrow, wptr + (size_t)ncol * INTER, hi, acc);

  #pragma unroll
  for (int r = 0; r < 8; ++r) {
    const int m = r + hi * 8;
    if (m < mlimit) {
      const int tok = tlds[m];
      const float c = (e < NEXP) ? comb[tok * NEXP + e] : 1.f;
      float* yrow = y + (size_t)tok * DIM;
      #pragma unroll
      for (int j = 0; j < 8; ++j)
        unsafeAtomicAdd(yrow + ncol + j * 128, c * acc[j][r]);
    }
  }
}

// ---------------------------------------------------------------------------
// Launch. Workspace layout (needs ~10.6 MB):
//   [0,   64K)  comb   float [1024][16]
//   [64K, 128K) list   int   [16][1024]
//   [128K,128K+512) offs int [17]
//   [+512, +8M)  Hrt   float [4096][512]
//   [..,  +2M)   Hsh   float [1024][512]
// ---------------------------------------------------------------------------
extern "C" void kernel_launch(void* const* d_in, const int* in_sizes, int n_in,
                              void* d_out, int out_size, void* d_ws, size_t ws_size,
                              hipStream_t stream) {
  const float* x    = (const float*)d_in[0];
  // d_in[1] = input_ids (unused by the reference computation)
  const float* gw   = (const float*)d_in[2];
  const float* gb   = (const float*)d_in[3];
  const float* w13  = (const float*)d_in[4];
  const float* w2   = (const float*)d_in[5];
  const float* w13s = (const float*)d_in[6];
  const float* w2s  = (const float*)d_in[7];
  float* y = (float*)d_out;

  char* ws = (char*)d_ws;
  float* comb = (float*)(ws);
  int*   list = (int*)(ws + (64 << 10));
  int*   offs = (int*)(ws + (128 << 10));
  float* Hrt  = (float*)(ws + (128 << 10) + 512);
  float* Hsh  = (float*)(ws + (128 << 10) + 512 + (size_t)TOPK * TOKENS * INTER * 4);

  zero_f32<<<(out_size + 255) / 256, 256, 0, stream>>>(y, out_size);
  moe_gate<<<TOKENS / 8, 256, 0, stream>>>(x, gw, gb, comb);
  moe_lists<<<1, 512, 0, stream>>>(comb, list, offs);
  dim3 g(TOKENS / 16, NEXP + 1);
  moe_gemm1<<<g, 256, 0, stream>>>(x, w13, w13s, list, offs, Hrt, Hsh);
  moe_gemm2<<<g, 256, 0, stream>>>(Hrt, Hsh, w2, w2s, list, offs, comb, y);
}
